// KacLayer_72688026517801
// MI455X (gfx1250) — matrix operand order repaired and checked by reading, not compiled, not run on hardware
//
#include <hip/hip_runtime.h>
#include <stdint.h>

// ============================================================================
// out = x @ (W^T + A1 * diag(vec) * A2) + b,  where A1/A2 are the Kac-walk
// rotation products reconstructed from numpy's default_rng(4048/4049).
// ============================================================================

#define DIM     1024
#define NSTEPS  3072          // ceil(log2(1024)*0.3) * 1024
#define NROWS   32768         // 8 * 4096

typedef _Float16 v8h  __attribute__((ext_vector_type(8)));
typedef _Float16 v16h __attribute__((ext_vector_type(16)));
typedef float    v8f  __attribute__((ext_vector_type(8)));

union HFrag { v16h v; v8h h8[2]; };

// ---------------------------------------------------------------------------
// numpy-compatible PCG64 (SeedSequence seeding, XSL-RR output, 32-bit buffer)
// ---------------------------------------------------------------------------
typedef unsigned __int128 u128;
__device__ __forceinline__ u128 pcg_mult() {
  return ((u128)0x2360ed051fc65da4ULL << 64) | (u128)0x4385df649fccf645ULL;
}

struct Pcg64 { u128 state; u128 inc; int has32; uint32_t buf32; };

// numpy SeedSequence(seed).generate_state(4, uint64)  (pool_size=4, 1 entropy word)
__device__ void seedseq_generate(uint32_t entropy0, uint64_t out64[4]) {
  const uint32_t XSHIFT = 16u;
  uint32_t pool[4];
  uint32_t hc = 0x43b0d7e5u;                    // INIT_A
  for (int i = 0; i < 4; i++) {
    uint32_t v = (i == 0) ? entropy0 : 0u;
    v ^= hc; hc *= 0x931e8875u; v *= hc; v ^= v >> XSHIFT;   // hashmix
    pool[i] = v;
  }
  for (int isrc = 0; isrc < 4; isrc++)
    for (int idst = 0; idst < 4; idst++)
      if (isrc != idst) {
        uint32_t v = pool[isrc];
        v ^= hc; hc *= 0x931e8875u; v *= hc; v ^= v >> XSHIFT;
        uint32_t r = (pool[idst] * 0xca01f9ddu) ^ (v * 0x4973f715u);
        r ^= r >> XSHIFT;
        pool[idst] = r;
      }
  uint32_t gh = 0x8b51f9ddu;                    // INIT_B
  uint32_t w32[8];
  for (int i = 0; i < 8; i++) {
    uint32_t d = pool[i & 3];
    d ^= gh; gh *= 0x58f38dedu; d *= gh; d ^= d >> XSHIFT;
    w32[i] = d;
  }
  for (int k = 0; k < 4; k++)
    out64[k] = (uint64_t)w32[2 * k] | ((uint64_t)w32[2 * k + 1] << 32);
}

__device__ void pcg64_init(Pcg64& g, uint32_t seedval) {
  uint64_t s[4]; seedseq_generate(seedval, s);
  u128 initstate = ((u128)s[0] << 64) | s[1];
  u128 initseq   = ((u128)s[2] << 64) | s[3];
  g.inc   = (initseq << 1) | 1;
  g.state = g.inc;                       // step from 0
  g.state += initstate;
  g.state = g.state * pcg_mult() + g.inc;
  g.has32 = 0; g.buf32 = 0;
}

__device__ __forceinline__ uint64_t pcg64_next64(Pcg64& g) {
  g.state = g.state * pcg_mult() + g.inc;
  uint64_t hi = (uint64_t)(g.state >> 64), lo = (uint64_t)g.state;
  uint64_t x  = hi ^ lo;
  unsigned rot = (unsigned)(hi >> 58);
  return (x >> rot) | (x << ((64u - rot) & 63u));
}

__device__ __forceinline__ uint32_t pcg64_next32(Pcg64& g) {
  if (g.has32) { g.has32 = 0; return g.buf32; }
  uint64_t n = pcg64_next64(g);
  g.has32 = 1; g.buf32 = (uint32_t)(n >> 32);
  return (uint32_t)n;
}

// LCG jump-ahead: lets every thread generate an independent slice of the stream
__device__ void pcg64_advance(Pcg64& g, uint64_t delta) {
  u128 acc_mult = 1, acc_plus = 0;
  u128 cur_mult = pcg_mult(), cur_plus = g.inc;
  while (delta) {
    if (delta & 1) { acc_mult *= cur_mult; acc_plus = acc_plus * cur_mult + cur_plus; }
    cur_plus = (cur_mult + 1) * cur_plus;
    cur_mult *= cur_mult;
    delta >>= 1;
  }
  g.state = acc_mult * g.state + acc_plus;
}

// numpy buffered_bounded_lemire_uint32 (Generator.integers path, use_masked=0)
__device__ __forceinline__ uint32_t lemire_u32(Pcg64& g, uint32_t rng /*incl max*/) {
  const uint32_t rng_excl = rng + 1u;
  uint64_t m = (uint64_t)pcg64_next32(g) * (uint64_t)rng_excl;
  uint32_t leftover = (uint32_t)m;
  if (leftover < rng_excl) {
    const uint32_t threshold = (0xFFFFFFFFu - rng) % rng_excl;
    while (leftover < threshold) {               // reject prob ~1e-9: kept for fidelity
      m = (uint64_t)pcg64_next32(g) * (uint64_t)rng_excl;
      leftover = (uint32_t)m;
    }
  }
  return (uint32_t)(m >> 32);
}

// ---------------------------------------------------------------------------
// Kernel 1: build walk tables (ii, jj, cos, sin) for both walks into ws.
// ii consumes u64 [0,1536), jj [1536,3072), theta [3072,6144). 12 draws/thread.
// ---------------------------------------------------------------------------
__global__ void rng_walk_tables(float* __restrict__ wsf) {
  const int w = blockIdx.x;                 // walk 0 (seed 4048) / walk 1 (4049)
  const int t = threadIdx.x;                // 0..255
  int*   ii = (int*)wsf + (size_t)w * 4 * NSTEPS;
  int*   jj = ii + NSTEPS;
  float* cs = wsf + (size_t)w * 4 * NSTEPS + 2 * NSTEPS;
  float* sn = cs + NSTEPS;

  Pcg64 base; pcg64_init(base, 4048u + (uint32_t)w);
  const int PER = NSTEPS / 256;             // 12
  const int U64 = PER / 2;                  // 6

  int myii[PER];
  { Pcg64 g = base; pcg64_advance(g, (uint64_t)(U64 * t));
    for (int q = 0; q < PER; q++) myii[q] = (int)lemire_u32(g, 1023u); }
  { Pcg64 g = base; pcg64_advance(g, (uint64_t)(NSTEPS / 2 + U64 * t));
    for (int q = 0; q < PER; q++) {
      uint32_t off = 1u + lemire_u32(g, 1022u);
      int s = t * PER + q;
      ii[s] = myii[q];
      jj[s] = (int)(((uint32_t)myii[q] + off) & (DIM - 1));
    } }
  { Pcg64 g = base; pcg64_advance(g, (uint64_t)(NSTEPS + PER * t));
    for (int q = 0; q < PER; q++) {
      uint64_t u = pcg64_next64(g);
      double nd = (double)(u >> 11) * (1.0 / 9007199254740992.0);
      double th = 6.283185307179586476925287 * nd;       // matches np 2*pi double
      int s = t * PER + q;
      cs[s] = (float)cos(th);
      sn[s] = (float)sin(th);
    } }
}

// ---------------------------------------------------------------------------
// Kernel 2: apply a walk to 32 identity rows held in LDS (one wave per block,
// lane == row, rotate-swizzled columns -> bank-conflict-free). Walk 0 emits
// As[d][k] = A1[d][k]*vec[k]; walk 1 emits Bt[e][k] = A2[k][e] (transposed).
// ---------------------------------------------------------------------------
__global__ __launch_bounds__(32) void kac_apply_identity(
    const float* __restrict__ wsf, const float* __restrict__ vec,
    float* __restrict__ outA, float* __restrict__ outBt) {
  extern __shared__ float lds[];            // 32 * 1024 floats = 128 KB dynamic
  const int blk  = blockIdx.x;              // 0..63
  const int w    = blk >> 5;
  const int grp  = blk & 31;
  const int lane = threadIdx.x;             // wave32: 0..31

  const int*   ii = (const int*)wsf + (size_t)w * 4 * NSTEPS;
  const int*   jj = ii + NSTEPS;
  const float* cs = wsf + (size_t)w * 4 * NSTEPS + 2 * NSTEPS;
  const float* sn = cs + NSTEPS;

  const int d = grp * 32 + lane;
  const unsigned rb = (unsigned)lane * DIM;
  for (int c = 0; c < DIM; c++) lds[rb + c] = 0.0f;
  lds[rb + ((unsigned)(d + lane) & (DIM - 1))] = 1.0f;
  // single wave, each lane owns its row, LDS in-order per wave: no barriers.

  for (int s = 0; s < NSTEPS; s++) {
    const int   i = ii[s], j = jj[s];
    const float c = cs[s], sv = sn[s];
    const unsigned pi = rb + ((unsigned)(i + lane) & (DIM - 1));
    const unsigned pj = rb + ((unsigned)(j + lane) & (DIM - 1));
    const float xi = lds[pi], xj = lds[pj];
    lds[pi] = c * xi - sv * xj;
    lds[pj] = sv * xi + c * xj;
  }

  if (w == 0) {  // As row-major, lanes stride columns -> coalesced
    for (int r2 = 0; r2 < 32; r2++) {
      const int dd = grp * 32 + r2;
      const unsigned rb2 = (unsigned)r2 * DIM;
      for (int c0 = 0; c0 < DIM; c0 += 32) {
        const int c = c0 + lane;
        outA[(size_t)dd * DIM + c] =
            lds[rb2 + ((unsigned)(c + r2) & (DIM - 1))] * vec[c];
      }
    }
  } else {       // Bt[e][d]: lanes provide consecutive d -> coalesced 128B
    for (int c = 0; c < DIM; c++) {
      outBt[(size_t)c * DIM + grp * 32 + lane] =
          lds[rb + ((unsigned)(c + lane) & (DIM - 1))];
    }
  }
}

// ---------------------------------------------------------------------------
// Kernel 3: split-f16 WMMA GEMM:  out[m][n] = sum_k A[m][k]*B[n][k] (+W)(+bias)
// fp32-class accuracy via (Ah*Bh + Al*Bh + Ah*Bl), V_WMMA_F32_16X16X32_F16.
// Block tile 128x128, 8 waves, each wave 2x4 16x16 accumulators.
// ---------------------------------------------------------------------------
#define BM 128
#define BN 128
#define BK 32
#define LDH 40   // half pitch: 80B rows keep 16B alignment, rotate LDS banks

__device__ __forceinline__ v8f wmma_f16(v16h a, v16h b, v8f c) {
  return __builtin_amdgcn_wmma_f32_16x16x32_f16(false, a, false, b,
                                                (short)0, c, false, false);
}

__global__ __launch_bounds__(256) void gemm_split_f16(
    const float* __restrict__ A,       // [Mdim][1024] row-major (K contiguous)
    const float* __restrict__ B,       // [1024][1024]: B[n][k]
    const float* __restrict__ addmat,  // optional [Mdim][1024] additive term
    const float* __restrict__ bias,    // optional [1024]
    float* __restrict__ out) {
  __shared__ _Float16 Ah[BM][LDH], Al[BM][LDH], Bh[BN][LDH], Bl[BN][LDH];
  const int tid  = threadIdx.x;
  const int lane = tid & 31;
  const int wave = tid >> 5;
  const int wm   = wave >> 1;                  // 0..3 -> 32-row slab
  const int wn   = wave & 1;                   // 0..1 -> 64-col slab
  const size_t m0 = (size_t)blockIdx.x * BM;
  const int    n0 = blockIdx.y * BN;
  const int l15 = lane & 15;
  const int lhi = (lane >> 4) & 1;

  v8f acc[2][4] = {};

  for (int k0 = 0; k0 < DIM; k0 += BK) {
    // ---- stage + hi/lo split: 128x32 f32 for A and B tiles ----
#pragma unroll
    for (int q = 0; q < 4; q++) {
      const int idx = q * 256 + tid;           // 0..1023 float4 slots
      const int row = idx >> 3;
      const int c4  = (idx & 7) * 4;
      const float4 av = *(const float4*)(A + (m0 + row) * DIM + k0 + c4);
      const float4 bv = *(const float4*)(B + (size_t)(n0 + row) * DIM + k0 + c4);
      const float af[4] = {av.x, av.y, av.z, av.w};
      const float bf[4] = {bv.x, bv.y, bv.z, bv.w};
#pragma unroll
      for (int e = 0; e < 4; e++) {
        _Float16 h = (_Float16)af[e];
        Ah[row][c4 + e] = h;
        Al[row][c4 + e] = (_Float16)(af[e] - (float)h);
        h = (_Float16)bf[e];
        Bh[row][c4 + e] = h;
        Bl[row][c4 + e] = (_Float16)(bf[e] - (float)h);
      }
    }
    __syncthreads();

    // ---- fragments per ISA layout:
    // A (16x32 f16): lanes<16 halves 0-7=K0-7, 8-15=K16-23; lanes>=16 +8.
    // B (32x16 f16): same K order as A: lanes<16 halves 0-7=K0-7, 8-15=K16-23; lanes>=16 +8.
    HFrag afh[2], afl[2];
#pragma unroll
    for (int mt = 0; mt < 2; mt++) {
      const int r  = wm * 32 + mt * 16 + l15;
      const int ko = lhi * 8;
      afh[mt].h8[0] = *(const v8h*)&Ah[r][ko];
      afh[mt].h8[1] = *(const v8h*)&Ah[r][16 + ko];
      afl[mt].h8[0] = *(const v8h*)&Al[r][ko];
      afl[mt].h8[1] = *(const v8h*)&Al[r][16 + ko];
    }
#pragma unroll
    for (int nt = 0; nt < 4; nt++) {
      const int cn = wn * 64 + nt * 16 + l15;
      const int ko = lhi * 8;
      HFrag bfh, bfl;
      bfh.h8[0] = *(const v8h*)&Bh[cn][ko];
      bfh.h8[1] = *(const v8h*)&Bh[cn][16 + ko];
      bfl.h8[0] = *(const v8h*)&Bl[cn][ko];
      bfl.h8[1] = *(const v8h*)&Bl[cn][16 + ko];
#pragma unroll
      for (int mt = 0; mt < 2; mt++) {
        acc[mt][nt] = wmma_f16(afh[mt].v, bfh.v, acc[mt][nt]);
        acc[mt][nt] = wmma_f16(afl[mt].v, bfh.v, acc[mt][nt]);
        acc[mt][nt] = wmma_f16(afh[mt].v, bfl.v, acc[mt][nt]);
      }
    }
    __syncthreads();
  }

  // ---- epilogue: C/D layout: VGPR v -> row v (+8 for hi lanes), col = lane&15
#pragma unroll
  for (int mt = 0; mt < 2; mt++)
#pragma unroll
    for (int nt = 0; nt < 4; nt++)
#pragma unroll
      for (int v = 0; v < 8; v++) {
        const size_t row = m0 + wm * 32 + mt * 16 + lhi * 8 + v;
        const int    col = n0 + wn * 64 + nt * 16 + l15;
        float x = acc[mt][nt][v];
        if (bias)   x += bias[col];
        if (addmat) x += addmat[row * DIM + col];
        out[row * DIM + col] = x;
      }
}

// ---------------------------------------------------------------------------
extern "C" void kernel_launch(void* const* d_in, const int* in_sizes, int n_in,
                              void* d_out, int out_size, void* d_ws, size_t ws_size,
                              hipStream_t stream) {
  (void)in_sizes; (void)n_in; (void)out_size; (void)ws_size;
  const float* x   = (const float*)d_in[0];   // [8,4096,1024] f32
  const float* W   = (const float*)d_in[1];   // [1024,1024]   f32 (W[e][d])
  const float* b   = (const float*)d_in[2];   // [1024]
  const float* vec = (const float*)d_in[3];   // [1024]
  float* out = (float*)d_out;
  float* wsf = (float*)d_ws;

  // ws: [walk tables 96KB][Mt 4MB]; d_out head reused as As/Bt scratch (8MB),
  // dead before the final GEMM overwrites all of d_out.
  float* As = out;                            // [1024][1024]  A1*diag(vec)
  float* Bt = out + (size_t)DIM * DIM;        // [1024][1024]  A2 transposed
  float* Mt = wsf + (size_t)2 * 4 * NSTEPS;   // [1024][1024]  (W^T + E)^T

  // walk kernel needs 128KB dynamic LDS (> default 64KB cap)
  (void)hipFuncSetAttribute(reinterpret_cast<const void*>(kac_apply_identity),
                            hipFuncAttributeMaxDynamicSharedMemorySize,
                            32 * DIM * 4);

  rng_walk_tables<<<dim3(2), dim3(256), 0, stream>>>(wsf);
  kac_apply_identity<<<dim3(64), dim3(32), 32 * DIM * 4, stream>>>(wsf, vec, As, Bt);
  // Mt[e][d] = W[e][d] + sum_k Bt[e][k] * As[d][k]
  gemm_split_f16<<<dim3(DIM / BM, DIM / BN), dim3(256), 0, stream>>>(
      Bt, As, W, nullptr, Mt);
  // out[r][e] = sum_d x[r][d] * Mt[e][d] + b[e]
  gemm_split_f16<<<dim3(NROWS / BM, DIM / BN), dim3(256), 0, stream>>>(
      x, Mt, nullptr, b, out);
}
